// GNNPolicy_4398046511886
// MI455X (gfx1250) — matrix operand order repaired and checked by reading, not compiled, not run on hardware
//
#include <hip/hip_runtime.h>
#include <math.h>

#define HDIM 128
#define NNODE 20000
#define NEDGE 640000
#define NLAYER 4
#define EDGE_WAVES 8   // 8 waves/block in the edge kernel -> 128 edges per block

typedef __attribute__((ext_vector_type(16))) __bf16 bf16x16;
typedef __attribute__((ext_vector_type(8)))  float  f32x8;

union BfVec { bf16x16 v; unsigned short us[16]; uint4 q[2]; };

__device__ __forceinline__ unsigned short f2bf(float f) {
  unsigned int u = __float_as_uint(f);
  u += 0x7fffu + ((u >> 16) & 1u);          // round-to-nearest-even
  return (unsigned short)(u >> 16);
}

__device__ __forceinline__ float eluf(float x) {
  return x > 0.f ? x : (__expf(x) - 1.f);
}

#define WMMA_BF16(a, b, c) \
  __builtin_amdgcn_wmma_f32_16x16x32_bf16(false, (a), false, (b), (short)0, (c), false, false)

#define ACC_ZERO {0.f,0.f,0.f,0.f,0.f,0.f,0.f,0.f}

// ---- fragment loaders (ISA 7.12.2 layouts) -------------------------------
// A (16x32 bf16): lane m = lane&15, khalf = lane>>4.
//   us[0..7]  <- K = s*32 + 8*khalf + {0..7}
//   us[8..15] <- K = s*32 + 16 + 8*khalf + {0..7}
__device__ __forceinline__ bf16x16 load_a_frag_bf(const unsigned short* A, int row0, int s) {
  int lane  = threadIdx.x & 31;
  int khalf = lane >> 4;
  const unsigned short* p = A + (size_t)(row0 + (lane & 15)) * HDIM + s * 32;
  BfVec u;
  u.q[0] = *(const uint4*)(p + 8 * khalf);
  u.q[1] = *(const uint4*)(p + 16 + 8 * khalf);
  return u.v;
}

__device__ __forceinline__ bf16x16 load_a_frag_f32(const float* A, int row0, int s) {
  int lane  = threadIdx.x & 31;
  int khalf = lane >> 4;
  const float* p = A + (size_t)(row0 + (lane & 15)) * HDIM + s * 32;
  BfVec u;
#pragma unroll
  for (int t = 0; t < 8; t++) u.us[t]     = f2bf(p[8 * khalf + t]);
#pragma unroll
  for (int t = 0; t < 8; t++) u.us[8 + t] = f2bf(p[16 + 8 * khalf + t]);
  return u.v;
}

// B fragments are pre-packed so each lane's 16 bf16 are contiguous (32B).
// Works for both global and (after inlining) LDS pointers.
__device__ __forceinline__ bf16x16 load_b_frag(const unsigned short* Bp, int njt, int j, int s) {
  int lane = threadIdx.x & 31;
  const unsigned short* p = Bp + ((size_t)((s * njt + j) * 32 + lane)) * 16;
  BfVec u;
  u.q[0] = ((const uint4*)p)[0];
  u.q[1] = ((const uint4*)p)[1];
  return u.v;
}

// ---- weight pre-pack: W[k][n] (KxN row-major f32) -> per-lane B fragments --
__global__ void k_pack_b(const float* __restrict__ W, unsigned short* __restrict__ out,
                         int njt, int total) {
  int idx = blockIdx.x * blockDim.x + threadIdx.x;
  if (idx >= total) return;
  int t    = idx & 15;
  int lane = (idx >> 4) & 31;
  int rest = idx >> 9;
  int j    = rest % njt;
  int s    = rest / njt;
  int n     = j * 16 + (lane & 15);
  int k     = s * 32 + (lane >> 4) * 8 + (t & 7) + (t >> 3) * 16;
  int ncols = njt * 16;
  out[idx] = f2bf(W[(size_t)k * ncols + n]);
}

__global__ void k_zero_f32(float* p, int total) {
  int idx = blockIdx.x * blockDim.x + threadIdx.x;
  if (idx < total) p[idx] = 0.f;
}

// ---- h = elu(x @ Wn + bn), keep f32 + bf16 copies -------------------------
__global__ void k_node_encode(const float* __restrict__ x, const float* __restrict__ Wn,
                              const float* __restrict__ bn,
                              float* __restrict__ h, unsigned short* __restrict__ hbf,
                              int total) {
  int idx = blockIdx.x * blockDim.x + threadIdx.x;
  if (idx >= total) return;
  int n = idx >> 7, j = idx & 127;
  const float* xr = x + (size_t)n * 4;
  float acc = bn[j];
#pragma unroll
  for (int k = 0; k < 4; k++) acc += xr[k] * Wn[k * HDIM + j];
  float v = eluf(acc);
  h[idx]   = v;
  hbf[idx] = f2bf(v);
}

// ---- edge scalar attr: a = ||pos[col]-pos[row]|| --------------------------
__global__ void k_edge_attr(const int* __restrict__ eidx, const float* __restrict__ pos,
                            float* __restrict__ a_edge) {
  int e = blockIdx.x * blockDim.x + threadIdx.x;
  if (e >= NEDGE) return;
  int r = eidx[e], c = eidx[NEDGE + e];
  float dx = pos[(size_t)c * 2]     - pos[(size_t)r * 2];
  float dy = pos[(size_t)c * 2 + 1] - pos[(size_t)r * 2 + 1];
  a_edge[e] = sqrtf(dx * dx + dy * dy + 1e-12f);
}

// ---- dual-output node GEMM: C1 = A@B1, C2 = A@B2 (shared A fragments) -----
__global__ void k_gemm128_dual(const unsigned short* __restrict__ Abf,
                               const unsigned short* __restrict__ Bp1,
                               const unsigned short* __restrict__ Bp2,
                               float* __restrict__ C1, float* __restrict__ C2, int mtiles) {
  int tile = blockIdx.x * (blockDim.x >> 5) + (threadIdx.x >> 5);
  if (tile >= mtiles) return;
  int row0  = tile * 16;
  int lane  = threadIdx.x & 31;
  int n     = lane & 15;
  int khalf = lane >> 4;
  bf16x16 a[4];
#pragma unroll
  for (int s = 0; s < 4; s++) a[s] = load_a_frag_bf(Abf, row0, s);
#pragma unroll
  for (int j = 0; j < 8; j++) {
    f32x8 acc1 = ACC_ZERO, acc2 = ACC_ZERO;
#pragma unroll
    for (int s = 0; s < 4; s++) acc1 = WMMA_BF16(a[s], load_b_frag(Bp1, 8, j, s), acc1);
#pragma unroll
    for (int s = 0; s < 4; s++) acc2 = WMMA_BF16(a[s], load_b_frag(Bp2, 8, j, s), acc2);
#pragma unroll
    for (int i = 0; i < 8; i++) {
      size_t o = (size_t)(row0 + i + 8 * khalf) * HDIM + j * 16 + n;
      C1[o] = acc1[i];
      C2[o] = acc2[i];
    }
  }
}

// ---- fused edge message: stage Wm_e in LDS, regenerate e-tile in registers,
//      WMMA e@Wm_e, gather HS/HD, elu, atomic scatter into agg[col] ---------
__global__ __launch_bounds__(EDGE_WAVES * 32)
void k_edge_msg(const int* __restrict__ eidx, const float* __restrict__ a_edge,
                const float* __restrict__ We, const float* __restrict__ be,
                const unsigned short* __restrict__ Wme_p,
                const float* __restrict__ HS, const float* __restrict__ HD,
                const float* __restrict__ bm, float* __restrict__ agg) {
  __shared__ unsigned short Bs[HDIM * HDIM];       // 32 KB packed Wm_e, block-shared
  {
    const uint4* src = (const uint4*)Wme_p;
    uint4*       dst = (uint4*)Bs;
#pragma unroll
    for (int i = threadIdx.x; i < (HDIM * HDIM) / 8; i += EDGE_WAVES * 32)
      dst[i] = src[i];
  }
  __syncthreads();

  int tile = blockIdx.x * EDGE_WAVES + (threadIdx.x >> 5);
  if (tile * 16 >= NEDGE) return;
  int lane  = threadIdx.x & 31;
  int n     = lane & 15;
  int khalf = lane >> 4;

  int r = 0, c = 0; float a = 0.f;
  if (lane < 16) {
    int eid = tile * 16 + lane;
    r = eidx[eid];
    c = eidx[NEDGE + eid];
    a = a_edge[eid];
  }
  float am = __shfl(a, n, 32);   // a of this lane's A-row m = lane&15

  // Regenerate e[m][k] = elu(a_m*We[k]+be[k]) directly in A-fragment layout.
  bf16x16 af[4];
#pragma unroll
  for (int s = 0; s < 4; s++) {
    BfVec u;
#pragma unroll
    for (int t = 0; t < 8; t++) {
      int k0 = s * 32 + 8 * khalf + t;
      int k1 = s * 32 + 16 + 8 * khalf + t;
      u.us[t]     = f2bf(eluf(am * We[k0] + be[k0]));
      u.us[8 + t] = f2bf(eluf(am * We[k1] + be[k1]));
    }
    af[s] = u.v;
  }

#pragma unroll
  for (int j = 0; j < 8; j++) {
    f32x8 acc = ACC_ZERO;
#pragma unroll
    for (int s = 0; s < 4; s++) acc = WMMA_BF16(af[s], load_b_frag(Bs, 8, j, s), acc);
    int colg = j * 16 + n;
    float bmv = bm[colg];
#pragma unroll
    for (int i = 0; i < 8; i++) {
      int m  = i + 8 * khalf;
      int rI = __shfl(r, m, 32);
      int cI = __shfl(c, m, 32);
      float v = acc[i] + HS[(size_t)rI * HDIM + colg] + HD[(size_t)cI * HDIM + colg] + bmv;
      atomicAdd(&agg[(size_t)cI * HDIM + colg], eluf(v));
    }
  }
}

// ---- node update: h = h + elu(h@Wu_h + agg@Wu_agg + bu)  (dual WMMA GEMM) -
__global__ void k_node_update(unsigned short* __restrict__ hbf, const float* __restrict__ agg,
                              const unsigned short* __restrict__ Wuh_p,
                              const unsigned short* __restrict__ Wua_p,
                              const float* __restrict__ bu,
                              float* __restrict__ h, int mtiles) {
  int tile = blockIdx.x * (blockDim.x >> 5) + (threadIdx.x >> 5);
  if (tile >= mtiles) return;
  int row0  = tile * 16;
  int lane  = threadIdx.x & 31;
  int n     = lane & 15;
  int khalf = lane >> 4;
  bf16x16 a1[4], a2[4];
#pragma unroll
  for (int s = 0; s < 4; s++) a1[s] = load_a_frag_bf(hbf, row0, s);
#pragma unroll
  for (int s = 0; s < 4; s++) a2[s] = load_a_frag_f32(agg, row0, s);
#pragma unroll
  for (int j = 0; j < 8; j++) {
    f32x8 acc = ACC_ZERO;
#pragma unroll
    for (int s = 0; s < 4; s++) acc = WMMA_BF16(a1[s], load_b_frag(Wuh_p, 8, j, s), acc);
#pragma unroll
    for (int s = 0; s < 4; s++) acc = WMMA_BF16(a2[s], load_b_frag(Wua_p, 8, j, s), acc);
#pragma unroll
    for (int i = 0; i < 8; i++) {
      size_t o = (size_t)(row0 + i + 8 * khalf) * HDIM + j * 16 + n;
      float hn = h[o] + eluf(acc[i] + bu[j * 16 + n]);
      h[o]   = hn;
      hbf[o] = f2bf(hn);    // safe: this wave's A frags already in registers
    }
  }
}

// ---- topo head: score = elu(h@Wt1+bt1) @ Wt2 + bt2 ------------------------
__global__ void k_topo(const unsigned short* __restrict__ hbf,
                       const unsigned short* __restrict__ Wt1p,
                       const float* __restrict__ bt1, const float* __restrict__ Wt2,
                       const float* __restrict__ bt2, float* __restrict__ out, int mtiles) {
  int tile = blockIdx.x * (blockDim.x >> 5) + (threadIdx.x >> 5);
  if (tile >= mtiles) return;
  int row0  = tile * 16;
  int lane  = threadIdx.x & 31;
  int n     = lane & 15;
  int khalf = lane >> 4;
  bf16x16 a[4];
#pragma unroll
  for (int s = 0; s < 4; s++) a[s] = load_a_frag_bf(hbf, row0, s);
  float part[8] = {0.f,0.f,0.f,0.f,0.f,0.f,0.f,0.f};
#pragma unroll
  for (int j = 0; j < 4; j++) {
    f32x8 acc = ACC_ZERO;
#pragma unroll
    for (int s = 0; s < 4; s++) acc = WMMA_BF16(a[s], load_b_frag(Wt1p, 4, j, s), acc);
    int colg = j * 16 + n;
    float w2 = Wt2[colg], b1 = bt1[colg];
#pragma unroll
    for (int i = 0; i < 8; i++) part[i] += eluf(acc[i] + b1) * w2;
  }
  // reduce over the 16 lanes within each khalf group
#pragma unroll
  for (int mask = 8; mask >= 1; mask >>= 1)
#pragma unroll
    for (int i = 0; i < 8; i++) part[i] += __shfl_xor(part[i], mask, 32);
  if (n == 0) {
    float b2 = bt2[0];
#pragma unroll
    for (int i = 0; i < 8; i++) out[row0 + 8 * khalf + i] = part[i] + b2;
  }
}

// ---- CVAE head (tiny, single block) ---------------------------------------
__global__ void k_head(const float* __restrict__ h, const float* __restrict__ z_c,
                       const float* __restrict__ tc,
                       const int* u_idx, const int* v_idx, const int* w_idx,
                       const float* Ge1, const float* ge1b, const float* Ge2, const float* ge2b,
                       const float* Gmu, const float* gmub, const float* Glv, const float* glvb,
                       const float* Gd1, const float* gd1b, const float* Gd2, const float* gd2b,
                       const float* Gd3, const float* gd3b, const float* Gp1, const float* gp1b,
                       const float* Gp2, const float* gp2b, const float* pbs,
                       float* __restrict__ out) {
  __shared__ float cond[256], xin[320], s1[128], s2[64], smu[64], sp[64];
  int tid = threadIdx.x;
  int u = u_idx[0], v = v_idx[0], w = w_idx[0];
  if (tid < 128)
    cond[tid] = (h[(size_t)u*HDIM+tid] + h[(size_t)v*HDIM+tid] + h[(size_t)w*HDIM+tid]) * (1.f/3.f);
  else
    cond[tid] = z_c[tid - 128];
  __syncthreads();
  if (tid < 4) xin[tid] = tc[tid];
  xin[4 + tid] = cond[tid];                       // enc input [260]
  __syncthreads();
  if (tid < 128) {
    float acc = ge1b[tid];
    for (int k = 0; k < 260; k++) acc += xin[k] * Ge1[k*128 + tid];
    s1[tid] = eluf(acc);
  }
  __syncthreads();
  if (tid < 64) {
    float acc = ge2b[tid];
    for (int k = 0; k < 128; k++) acc += s1[k] * Ge2[k*64 + tid];
    s2[tid] = eluf(acc);
  }
  __syncthreads();
  if (tid < 64) {
    float mu = gmub[tid], lv = glvb[tid];
    for (int k = 0; k < 64; k++) { mu += s2[k]*Gmu[k*64+tid]; lv += s2[k]*Glv[k*64+tid]; }
    smu[tid] = mu;
    out[NNODE + 4 + tid]  = mu;
    out[NNODE + 68 + tid] = lv;
  }
  __syncthreads();
  if (tid < 64) xin[tid] = smu[tid];
  xin[64 + tid] = cond[tid];                      // dec input [320]
  __syncthreads();
  if (tid < 128) {
    float acc = gd1b[tid];
    for (int k = 0; k < 320; k++) acc += xin[k] * Gd1[k*128 + tid];
    s1[tid] = eluf(acc);
  } else if (tid < 192) {
    int j = tid - 128;
    float acc = gp1b[j];
    for (int k = 0; k < 256; k++) acc += cond[k] * Gp1[k*64 + j];
    sp[j] = eluf(acc);
  }
  __syncthreads();
  if (tid < 64) {
    float acc = gd2b[tid];
    for (int k = 0; k < 128; k++) acc += s1[k] * Gd2[k*64 + tid];
    s2[tid] = eluf(acc);
  }
  __syncthreads();
  if (tid < 4) {
    float base = gd3b[tid], pb = gp2b[tid];
    for (int k = 0; k < 64; k++) { base += s2[k]*Gd3[k*4+tid]; pb += sp[k]*Gp2[k*4+tid]; }
    float sc = fminf(fmaxf(pbs[0], 0.f), 0.5f);
    out[NNODE + tid] = base + tanhf(pb) * sc;
  }
}

extern "C" void kernel_launch(void* const* d_in, const int* in_sizes, int n_in,
                              void* d_out, int out_size, void* d_ws, size_t ws_size,
                              hipStream_t stream) {
  const float* x      = (const float*)d_in[0];
  const float* pos    = (const float*)d_in[1];
  const int*   eidx   = (const int*)  d_in[2];
  const float* z_c    = (const float*)d_in[3];
  const float* tc     = (const float*)d_in[4];
  const int*   u_idx  = (const int*)  d_in[5];
  const int*   v_idx  = (const int*)  d_in[6];
  const int*   w_idx  = (const int*)  d_in[7];
  const float* Wn     = (const float*)d_in[8];
  const float* bn     = (const float*)d_in[9];
  const float* We     = (const float*)d_in[10];
  const float* be     = (const float*)d_in[11];
  const float* Wm_src = (const float*)d_in[12];
  const float* Wm_dst = (const float*)d_in[13];
  const float* Wm_e   = (const float*)d_in[14];
  const float* bm     = (const float*)d_in[15];
  const float* Wu_h   = (const float*)d_in[16];
  const float* Wu_agg = (const float*)d_in[17];
  const float* bu     = (const float*)d_in[18];
  const float* Wt1    = (const float*)d_in[19];
  const float* bt1    = (const float*)d_in[20];
  const float* Wt2    = (const float*)d_in[21];
  const float* bt2    = (const float*)d_in[22];
  const float* Ge1    = (const float*)d_in[23];
  const float* ge1b   = (const float*)d_in[24];
  const float* Ge2    = (const float*)d_in[25];
  const float* ge2b   = (const float*)d_in[26];
  const float* Gmu    = (const float*)d_in[27];
  const float* gmub   = (const float*)d_in[28];
  const float* Glv    = (const float*)d_in[29];
  const float* glvb   = (const float*)d_in[30];
  const float* Gd1    = (const float*)d_in[31];
  const float* gd1b   = (const float*)d_in[32];
  const float* Gd2    = (const float*)d_in[33];
  const float* gd2b   = (const float*)d_in[34];
  const float* Gd3    = (const float*)d_in[35];
  const float* gd3b   = (const float*)d_in[36];
  const float* Gp1    = (const float*)d_in[37];
  const float* gp1b   = (const float*)d_in[38];
  const float* Gp2    = (const float*)d_in[39];
  const float* gp2b   = (const float*)d_in[40];
  const float* pbs    = (const float*)d_in[41];

  float* out = (float*)d_out;

  // workspace carve-up (256B aligned)
  char* ws = (char*)d_ws;
  size_t off = 0;
  auto take = [&](size_t bytes) -> char* {
    char* p = ws + off;
    off += (bytes + 255) & ~(size_t)255;
    return p;
  };
  float*          h_f32  = (float*)         take((size_t)NNODE * HDIM * 4);
  unsigned short* h_bf   = (unsigned short*)take((size_t)NNODE * HDIM * 2);
  float*          HS     = (float*)         take((size_t)NNODE * HDIM * 4);
  float*          HD     = (float*)         take((size_t)NNODE * HDIM * 4);
  float*          agg    = (float*)         take((size_t)NNODE * HDIM * 4);
  float*          a_edge = (float*)         take((size_t)NEDGE * 4);
  unsigned short* packs  = (unsigned short*)take((size_t)(20 * 16384 + 8192) * 2);
  (void)ws_size; (void)n_in; (void)in_sizes; (void)out_size;

  unsigned short* wsrc_p[NLAYER], *wdst_p[NLAYER], *wme_p[NLAYER], *wuh_p[NLAYER], *wua_p[NLAYER];
  for (int l = 0; l < NLAYER; l++) {
    wsrc_p[l] = packs + (size_t)(0  + l) * 16384;
    wdst_p[l] = packs + (size_t)(4  + l) * 16384;
    wme_p[l]  = packs + (size_t)(8  + l) * 16384;
    wuh_p[l]  = packs + (size_t)(12 + l) * 16384;
    wua_p[l]  = packs + (size_t)(16 + l) * 16384;
  }
  unsigned short* wt1p = packs + (size_t)20 * 16384;

  // pre-pack weights into WMMA B-fragment layout (bf16)
  for (int l = 0; l < NLAYER; l++) {
    k_pack_b<<<64, 256, 0, stream>>>(Wm_src + (size_t)l * 16384, wsrc_p[l], 8, 16384);
    k_pack_b<<<64, 256, 0, stream>>>(Wm_dst + (size_t)l * 16384, wdst_p[l], 8, 16384);
    k_pack_b<<<64, 256, 0, stream>>>(Wm_e   + (size_t)l * 16384, wme_p[l],  8, 16384);
    k_pack_b<<<64, 256, 0, stream>>>(Wu_h   + (size_t)l * 16384, wuh_p[l],  8, 16384);
    k_pack_b<<<64, 256, 0, stream>>>(Wu_agg + (size_t)l * 16384, wua_p[l],  8, 16384);
  }
  k_pack_b<<<32, 256, 0, stream>>>(Wt1, wt1p, 4, 8192);

  const int node_elems = NNODE * HDIM;
  k_node_encode<<<(node_elems + 255) / 256, 256, 0, stream>>>(x, Wn, bn, h_f32, h_bf, node_elems);
  k_edge_attr<<<(NEDGE + 255) / 256, 256, 0, stream>>>(eidx, pos, a_edge);

  const int mtiles = NNODE / 16;                     // 1250
  const int gemm_blocks = (mtiles + 3) / 4;          // 4 waves/block
  const int edge_blocks = (NEDGE / 16) / EDGE_WAVES; // 5000 blocks, 8 waves each

  for (int l = 0; l < NLAYER; l++) {
    k_zero_f32<<<(node_elems + 255) / 256, 256, 0, stream>>>(agg, node_elems);
    k_gemm128_dual<<<gemm_blocks, 128, 0, stream>>>(h_bf, wsrc_p[l], wdst_p[l], HS, HD, mtiles);
    k_edge_msg<<<edge_blocks, EDGE_WAVES * 32, 0, stream>>>(eidx, a_edge, We, be, wme_p[l],
                                                            HS, HD, bm + (size_t)l * HDIM, agg);
    k_node_update<<<gemm_blocks, 128, 0, stream>>>(h_bf, agg, wuh_p[l], wua_p[l],
                                                   bu + (size_t)l * HDIM, h_f32, mtiles);
  }

  k_topo<<<gemm_blocks, 128, 0, stream>>>(h_bf, wt1p, bt1, Wt2, bt2, out, mtiles);
  k_head<<<1, 256, 0, stream>>>(h_f32, z_c, tc, u_idx, v_idx, w_idx,
                                Ge1, ge1b, Ge2, ge2b, Gmu, gmub, Glv, glvb,
                                Gd1, gd1b, Gd2, gd2b, Gd3, gd3b, Gp1, gp1b,
                                Gp2, gp2b, pbs, out);
}
